// DownSample_17145509446489
// MI455X (gfx1250) — matrix operand order repaired and checked by reading, not compiled, not run on hardware
//
#include <hip/hip_runtime.h>
#include <hip/hip_bf16.h>
#include <math.h>

// ---------------------------------------------------------------------------
// Problem constants (from reference): B=4, H=4, N=2048, C=128, D=32, K=1024
// ---------------------------------------------------------------------------
#define BATCH 4
#define HEADS 4
#define NPTS  2048
#define CHAN  128
#define DEPTH 32
#define KKEEP 1024
#define NBH   (BATCH * HEADS)

typedef __bf16 v16bf __attribute__((ext_vector_type(16)));
typedef float  v8f   __attribute__((ext_vector_type(8)));

union BF16x16 { v16bf v; unsigned int u[8]; };

__device__ __forceinline__ unsigned short f2bf(float f) {
    unsigned int u = __float_as_uint(f);
    unsigned int r = u + 0x7FFFu + ((u >> 16) & 1u);   // round-to-nearest-even
    return (unsigned short)(r >> 16);
}

__device__ __forceinline__ v8f wmma_bf16(const BF16x16 &a, const BF16x16 &b, v8f c) {
    // v_wmma_f32_16x16x32_bf16: D = A(16x32) * B(32x16) + C(16x16 f32)
    return __builtin_amdgcn_wmma_f32_16x16x32_bf16(
        /*neg_a=*/false, a.v, /*neg_b=*/false, b.v,
        /*c_mod=*/(short)0, c, /*reuse_a=*/false, /*reuse_b=*/false);
}

// K-pair start for A-operand register j, lane-half `half` (contiguous k = 16*half + 2*j, matches B fills)
__device__ __forceinline__ int a_k0(int j, int half) {
    return half * 16 + j * 2;
}

// cross-lane reductions within groups of 16 lanes (wave32) -- used ONCE per row
__device__ __forceinline__ float rmax16(float v) {
    v = fmaxf(v, __shfl_xor(v, 1, 16));
    v = fmaxf(v, __shfl_xor(v, 2, 16));
    v = fmaxf(v, __shfl_xor(v, 4, 16));
    v = fmaxf(v, __shfl_xor(v, 8, 16));
    return v;
}
__device__ __forceinline__ float rsum16(float v) {
    v += __shfl_xor(v, 1, 16);
    v += __shfl_xor(v, 2, 16);
    v += __shfl_xor(v, 4, 16);
    v += __shfl_xor(v, 8, 16);
    return v;
}

#define SM_SCALE 0.17677669529663687f  // 1/sqrt(32)

// ---------------------------------------------------------------------------
// K1: q/k/v projections (1x1 conv == channel matmul), f32 in -> bf16 out.
//   q,k stored [bh][n][d] (A/B operand friendly for energy)
//   v   stored [bh][d][n] (B operand friendly for attn @ v^T)
// ---------------------------------------------------------------------------
__global__ __launch_bounds__(128) void proj_kernel(
    const float *__restrict__ x, const float *__restrict__ wq,
    const float *__restrict__ wk, const float *__restrict__ wv,
    unsigned short *__restrict__ qb, unsigned short *__restrict__ kb,
    unsigned short *__restrict__ vb) {
    __shared__ float xcol[CHAN];
    const int blk = blockIdx.x;            // b*N + n
    const int n = blk & (NPTS - 1);
    const int b = blk >> 11;
    const int o = threadIdx.x;             // output channel 0..127
    xcol[o] = x[(b * CHAN + o) * NPTS + n];
    __syncthreads();
    float aq = 0.f, ak = 0.f, av = 0.f;
#pragma unroll 8
    for (int i = 0; i < CHAN; ++i) {
        const float xv = xcol[i];
        aq = fmaf(wq[o * CHAN + i], xv, aq);
        ak = fmaf(wk[o * CHAN + i], xv, ak);
        av = fmaf(wv[o * CHAN + i], xv, av);
    }
    const int h = o >> 5, d = o & 31;
    const int bh = b * HEADS + h;
    qb[(bh * NPTS + n) * DEPTH + d] = f2bf(aq);
    kb[(bh * NPTS + n) * DEPTH + d] = f2bf(ak);
    vb[(bh * DEPTH + d) * NPTS + n] = f2bf(av);
}

// ---------------------------------------------------------------------------
// K2: softmax row stats in two lane-local passes (max, then sumexp).
// Cross-lane reduction happens exactly once per row per pass.
// Writes stats[n] = (rowmax, 1/rowsum) as float2.
// ---------------------------------------------------------------------------
__global__ __launch_bounds__(32) void rowstats_kernel(
    const unsigned short *__restrict__ qb, const unsigned short *__restrict__ kb,
    float2 *__restrict__ stats) {
    const int lane = threadIdx.x;
    const int ntile = blockIdx.x & (NPTS / 16 - 1);
    const int bh = blockIdx.x >> 7;
    const int n0 = ntile * 16;
    const int half = lane >> 4, l16 = lane & 15;
    const unsigned int *qb32 = (const unsigned int *)qb;
    const unsigned int *kb32 = (const unsigned int *)kb;

    // A operand: q rows n0..n0+15 (hoisted, reused by both passes)
    BF16x16 A;
#pragma unroll
    for (int j = 0; j < 8; ++j)
        A.u[j] = qb32[(bh * NPTS + n0 + l16) * 16 + (a_k0(j, half) >> 1)];

    // ---- pass 1: lane-local row max ----
    float Ml[8];
#pragma unroll
    for (int r = 0; r < 8; ++r) Ml[r] = -3.0e38f;
    for (int m0 = 0; m0 < NPTS; m0 += 32) {
#pragma unroll
        for (int t = 0; t < 2; ++t) {
            const int mcol = m0 + t * 16 + l16;
            BF16x16 Bm;
#pragma unroll
            for (int j = 0; j < 8; ++j)
                Bm.u[j] = kb32[(bh * NPTS + mcol) * 16 + half * 8 + j];
            v8f c = {0.f, 0.f, 0.f, 0.f, 0.f, 0.f, 0.f, 0.f};
            v8f e = wmma_bf16(A, Bm, c);
#pragma unroll
            for (int r = 0; r < 8; ++r)
                Ml[r] = fmaxf(Ml[r], e[r] * SM_SCALE);
        }
    }
    float M[8];
#pragma unroll
    for (int r = 0; r < 8; ++r) M[r] = rmax16(Ml[r]);

    // ---- pass 2: lane-local sum of exp(E - M) ----
    float Sl[8];
#pragma unroll
    for (int r = 0; r < 8; ++r) Sl[r] = 0.f;
    for (int m0 = 0; m0 < NPTS; m0 += 32) {
#pragma unroll
        for (int t = 0; t < 2; ++t) {
            const int mcol = m0 + t * 16 + l16;
            BF16x16 Bm;
#pragma unroll
            for (int j = 0; j < 8; ++j)
                Bm.u[j] = kb32[(bh * NPTS + mcol) * 16 + half * 8 + j];
            v8f c = {0.f, 0.f, 0.f, 0.f, 0.f, 0.f, 0.f, 0.f};
            v8f e = wmma_bf16(A, Bm, c);
#pragma unroll
            for (int r = 0; r < 8; ++r)
                Sl[r] += __expf(e[r] * SM_SCALE - M[r]);
        }
    }
    if (true) {
#pragma unroll
        for (int r = 0; r < 8; ++r) {
            const float S = rsum16(Sl[r]);
            if (l16 == 0) {
                const int n = n0 + r + 8 * half;
                stats[bh * NPTS + n] = make_float2(M[r], 1.0f / S);
            }
        }
    }
}

// ---------------------------------------------------------------------------
// K3: vd[n,d] = attn[n,:] @ v[d,:]  (bf16 WMMA, f32 accum; LDS relayout of the
// attn tile from C/D layout to A-operand layout). No atomics, no colsum here.
// ---------------------------------------------------------------------------
__global__ __launch_bounds__(32) void attnvd_kernel(
    const unsigned short *__restrict__ qb, const unsigned short *__restrict__ kb,
    const unsigned short *__restrict__ vb, const float2 *__restrict__ stats,
    float *__restrict__ vdws) {
    __shared__ unsigned short sattn[16 * 32]; // attn strip, bf16, row-major [16][32]
    const int lane = threadIdx.x;
    const int ntile = blockIdx.x & (NPTS / 16 - 1);
    const int bh = blockIdx.x >> 7;
    const int n0 = ntile * 16;
    const int half = lane >> 4, l16 = lane & 15;
    const unsigned int *qb32 = (const unsigned int *)qb;
    const unsigned int *kb32 = (const unsigned int *)kb;
    const unsigned int *vb32 = (const unsigned int *)vb;
    const unsigned int *sattn32 = (const unsigned int *)sattn;

    BF16x16 A;
#pragma unroll
    for (int j = 0; j < 8; ++j)
        A.u[j] = qb32[(bh * NPTS + n0 + l16) * 16 + (a_k0(j, half) >> 1)];

    float Mrow[8], Zinv[8];
#pragma unroll
    for (int r = 0; r < 8; ++r) {
        const float2 st = stats[bh * NPTS + n0 + r + 8 * half];
        Mrow[r] = st.x;
        Zinv[r] = st.y;
    }

    v8f acc0 = {0.f, 0.f, 0.f, 0.f, 0.f, 0.f, 0.f, 0.f};
    v8f acc1 = acc0;

    for (int m0 = 0; m0 < NPTS; m0 += 32) {
        if (m0 + 32 < NPTS) // warm caches for next chunk of k
            __builtin_prefetch(&kb32[(bh * NPTS + m0 + 32 + l16) * 16], 0, 3);
#pragma unroll
        for (int t = 0; t < 2; ++t) {
            const int mcol = m0 + t * 16 + l16;
            BF16x16 Bm;
#pragma unroll
            for (int j = 0; j < 8; ++j)
                Bm.u[j] = kb32[(bh * NPTS + mcol) * 16 + half * 8 + j];
            v8f c = {0.f, 0.f, 0.f, 0.f, 0.f, 0.f, 0.f, 0.f};
            v8f e = wmma_bf16(A, Bm, c);
#pragma unroll
            for (int r = 0; r < 8; ++r) {
                const float a = __expf(e[r] * SM_SCALE - Mrow[r]) * Zinv[r];
                sattn[(r + 8 * half) * 32 + t * 16 + l16] = f2bf(a);
            }
        }
        __syncthreads();
        // reload attn strip in A-operand layout (16 rows x K=32)
        BF16x16 Aattn;
#pragma unroll
        for (int j = 0; j < 8; ++j)
            Aattn.u[j] = sattn32[l16 * 16 + (a_k0(j, half) >> 1)];
        // B operand: v^T tiles, K = m (32), cols = d (two 16-wide tiles)
#pragma unroll
        for (int t = 0; t < 2; ++t) {
            const int d = t * 16 + l16;
            BF16x16 Bv;
#pragma unroll
            for (int j = 0; j < 8; ++j) {
                const int mk = m0 + half * 16 + 2 * j;
                Bv.u[j] = vb32[(((bh * DEPTH + d) * NPTS) + mk) >> 1];
            }
            if (t == 0) acc0 = wmma_bf16(Aattn, Bv, acc0);
            else        acc1 = wmma_bf16(Aattn, Bv, acc1);
        }
        __syncthreads();
    }

#pragma unroll
    for (int r = 0; r < 8; ++r) {
        const int n = n0 + r + 8 * half;
        vdws[(bh * NPTS + n) * DEPTH + l16] = acc0[r];
        vdws[(bh * NPTS + n) * DEPTH + 16 + l16] = acc1[r];
    }
}

// ---------------------------------------------------------------------------
// K4: scores[m] = sum_n attn[n,m] via the TRANSPOSED energy (roles of q and k
// swapped: E^T[m,n] = k[m,:]·q[n,:]). Lane-local accumulation over all n,
// one cross-lane reduction per output. No atomics anywhere.
// ---------------------------------------------------------------------------
__global__ __launch_bounds__(32) void colsum_kernel(
    const unsigned short *__restrict__ qb, const unsigned short *__restrict__ kb,
    const float2 *__restrict__ stats, float *__restrict__ scores) {
    const int lane = threadIdx.x;
    const int mtile = blockIdx.x & (NPTS / 16 - 1);
    const int bh = blockIdx.x >> 7;
    const int m0 = mtile * 16;
    const int half = lane >> 4, l16 = lane & 15;
    const unsigned int *qb32 = (const unsigned int *)qb;
    const unsigned int *kb32 = (const unsigned int *)kb;

    // A operand: k rows m0..m0+15
    BF16x16 A;
#pragma unroll
    for (int j = 0; j < 8; ++j)
        A.u[j] = kb32[(bh * NPTS + m0 + l16) * 16 + (a_k0(j, half) >> 1)];

    float acc[8];
#pragma unroll
    for (int r = 0; r < 8; ++r) acc[r] = 0.f;

    for (int n0c = 0; n0c < NPTS; n0c += 32) {
#pragma unroll
        for (int t = 0; t < 2; ++t) {
            const int ncol = n0c + t * 16 + l16;
            BF16x16 Bq;
#pragma unroll
            for (int j = 0; j < 8; ++j)
                Bq.u[j] = qb32[(bh * NPTS + ncol) * 16 + half * 8 + j];
            v8f c = {0.f, 0.f, 0.f, 0.f, 0.f, 0.f, 0.f, 0.f};
            v8f e = wmma_bf16(A, Bq, c);           // e[m_r, n_col]
            const float2 st = stats[bh * NPTS + ncol]; // (M_n, 1/Z_n) of column n
#pragma unroll
            for (int r = 0; r < 8; ++r)
                acc[r] += __expf(e[r] * SM_SCALE - st.x) * st.y;
        }
    }
#pragma unroll
    for (int r = 0; r < 8; ++r) {
        const float v = rsum16(acc[r]);
        if (l16 == 0)
            scores[bh * NPTS + m0 + r + 8 * half] = v;
    }
}

// ---------------------------------------------------------------------------
// K5: exact jax top_k semantics via rank counting (ties -> lower index first)
// ---------------------------------------------------------------------------
__global__ __launch_bounds__(256) void topk_kernel(
    const float *__restrict__ scores, int *__restrict__ idx_out,
    int *__restrict__ idxd_out) {
    __shared__ float s[NPTS];
    const int bh = blockIdx.x >> 3;
    const int seg = blockIdx.x & 7;
    for (int i = threadIdx.x; i < NPTS; i += 256)
        s[i] = scores[bh * NPTS + i];
    __syncthreads();
    const int i = seg * 256 + threadIdx.x;
    const float si = s[i];
    int cg = 0, cl = 0;
    for (int j = 0; j < NPTS; ++j) {
        const float sj = s[j];
        const bool tie = (sj == si) && (j < i);
        cg += ((sj > si) || tie) ? 1 : 0;
        cl += ((sj < si) || tie) ? 1 : 0;
    }
    if (cg < KKEEP) idx_out[bh * KKEEP + cg] = i;        // descending rank
    if (cl < KKEEP) idxd_out[bh * KKEEP + cl] = i;       // ascending rank
}

// ---------------------------------------------------------------------------
// K6: gather vd rows per index set into (B, H*D, K) outputs
// ---------------------------------------------------------------------------
__global__ __launch_bounds__(256) void gather_kernel(
    const float *__restrict__ vdws, const int *__restrict__ idx_out,
    const int *__restrict__ idxd_out, float *__restrict__ out_down,
    float *__restrict__ out_drop) {
    const int total = BATCH * HEADS * DEPTH * KKEEP; // per selection
    int t = blockIdx.x * blockDim.x + threadIdx.x;
    if (t >= 2 * total) return;
    const int sel = (t >= total) ? 1 : 0;
    const int u = t - sel * total;
    const int j = u & (KKEEP - 1);
    const int d = (u >> 10) & (DEPTH - 1);
    const int h = (u >> 15) & (HEADS - 1);
    const int b = u >> 17;
    const int bh = b * HEADS + h;
    const int *idxs = sel ? idxd_out : idx_out;
    const int i = idxs[bh * KKEEP + j];
    const float val = vdws[(bh * NPTS + i) * DEPTH + d];
    float *o = sel ? out_drop : out_down;
    o[b * (CHAN * KKEEP) + (h * DEPTH + d) * KKEEP + j] = val;
}

// ---------------------------------------------------------------------------
extern "C" void kernel_launch(void *const *d_in, const int *in_sizes, int n_in,
                              void *d_out, int out_size, void *d_ws, size_t ws_size,
                              hipStream_t stream) {
    (void)in_sizes; (void)n_in; (void)out_size; (void)ws_size;
    const float *x  = (const float *)d_in[0];
    const float *wq = (const float *)d_in[1];
    const float *wk = (const float *)d_in[2];
    const float *wv = (const float *)d_in[3];

    // workspace layout
    char *ws = (char *)d_ws;
    unsigned short *qb = (unsigned short *)(ws);                     // 2 MiB
    unsigned short *kb = (unsigned short *)(ws + (2u << 20));        // 2 MiB
    unsigned short *vb = (unsigned short *)(ws + (4u << 20));        // 2 MiB
    float2 *stats = (float2 *)(ws + (6u << 20));                     // 256 KiB
    float  *scores = (float *)(ws + (6u << 20) + (256u << 10));      // 128 KiB
    float  *vdws   = (float *)(ws + (6u << 20) + (384u << 10));      // 4 MiB

    // output layout: v_down | idx | v_dropped | idx_dropped
    float *out = (float *)d_out;
    float *v_down   = out;                                           // 524288
    int   *idx_out  = (int *)(out + 524288);                         //  16384
    float *v_drop   = out + 540672;                                  // 524288
    int   *idxd_out = (int *)(out + 1064960);                        //  16384

    proj_kernel<<<BATCH * NPTS, 128, 0, stream>>>(x, wq, wk, wv, qb, kb, vb);
    rowstats_kernel<<<NBH * (NPTS / 16), 32, 0, stream>>>(qb, kb, stats);
    attnvd_kernel<<<NBH * (NPTS / 16), 32, 0, stream>>>(qb, kb, vb, stats, vdws);
    colsum_kernel<<<NBH * (NPTS / 16), 32, 0, stream>>>(qb, kb, stats, scores);
    topk_kernel<<<NBH * (NPTS / 256), 256, 0, stream>>>(scores, idx_out, idxd_out);
    const int gtot = 2 * BATCH * HEADS * DEPTH * KKEEP;
    gather_kernel<<<(gtot + 255) / 256, 256, 0, stream>>>(vdws, idx_out, idxd_out,
                                                          v_down, v_drop);
}
